// Attention_69260642615568
// MI455X (gfx1250) — matrix operand order repaired and checked by reading, not compiled, not run on hardware
//
#include <hip/hip_runtime.h>

#define BB 4
#define LL 2048
#define DD 1024
#define HH 16
#define DEP 64

typedef __attribute__((ext_vector_type(16))) __bf16 v16bf;
typedef __attribute__((ext_vector_type(8)))  float  v8f;

union AFrag { v16bf v; uint4 q[2]; };

// pack two floats -> two bf16 in one u32 via hw cvt (gfx1250 bf16 VALU)
__device__ __forceinline__ unsigned pk_bf16(float lo, float hi) {
    unsigned r;
    asm("v_cvt_pk_bf16_f32 %0, %1, %2" : "=v"(r) : "v"(lo), "v"(hi));
    return r;
}

__device__ __forceinline__ unsigned short f2bf(float f) {
    return (unsigned short)pk_bf16(f, f);
}

// ---------------------------------------------------------------------------
// Projection GEMM: Obf[b,h,l,dep] = bf16( scale * sum_k A[m,k] * W[k,n] )
// Block: 256 threads (8 waves), tile 128 rows x 64 cols, K-step 64.
// ---------------------------------------------------------------------------
__global__ __launch_bounds__(256) void proj_kernel(const float* __restrict__ A,
                                                   const float* __restrict__ W,
                                                   unsigned short* __restrict__ Obf,
                                                   float scale)
{
    __shared__ unsigned short ldsA[128 * 64];   // [row][k] bf16, 16KB
    __shared__ unsigned short ldsWt[64 * 64];   // transposed [n][k] bf16, 8KB

    const int tid  = threadIdx.x;
    const int lane = tid & 31;
    const int wave = tid >> 5;                  // 0..7
    const int rowBase = blockIdx.x * 128;
    const int colBase = blockIdx.y * 64;

    v8f acc[4] = {};

    for (int kb = 0; kb < DD; kb += 64) {
        __syncthreads();
        // stage A tile: 128x64 f32 -> bf16 (packed converts)
        #pragma unroll
        for (int i = 0; i < 8; ++i) {
            int idx = tid + i * 256;            // 2048 float4 total
            int r  = idx >> 4;                  // 16 float4 per row
            int c4 = idx & 15;
            const float4 g = *(const float4*)(A + (size_t)(rowBase + r) * DD + kb + c4 * 4);
            uint2 p;
            p.x = pk_bf16(g.x, g.y);
            p.y = pk_bf16(g.z, g.w);
            *(uint2*)&ldsA[r * 64 + c4 * 4] = p;
        }
        // stage W tile transposed: 64x64 -> ldsWt[n][k], u32 stores of k-pairs
        {
            unsigned* ldsWt32 = (unsigned*)ldsWt;
            #pragma unroll
            for (int i = 0; i < 2; ++i) {
                int idx = tid + i * 256;        // 512 work items
                int nq = idx & 15;              // n0 = nq*4
                int kp = idx >> 4;              // k0 = kp*2
                const float4 g0 = *(const float4*)(W + (size_t)(kb + kp * 2) * DD
                                                   + colBase + nq * 4);
                const float4 g1 = *(const float4*)(W + (size_t)(kb + kp * 2 + 1) * DD
                                                   + colBase + nq * 4);
                ldsWt32[(nq * 4 + 0) * 32 + kp] = pk_bf16(g0.x, g1.x);
                ldsWt32[(nq * 4 + 1) * 32 + kp] = pk_bf16(g0.y, g1.y);
                ldsWt32[(nq * 4 + 2) * 32 + kp] = pk_bf16(g0.z, g1.z);
                ldsWt32[(nq * 4 + 3) * 32 + kp] = pk_bf16(g0.w, g1.w);
            }
        }
        __syncthreads();

        // A fragments (16x32 each, two depth chunks)
        AFrag a[2];
        {
            int r = wave * 16 + (lane & 15);
            int off = (lane < 16) ? 0 : 16;     // lane-half K split
            #pragma unroll
            for (int dc = 0; dc < 2; ++dc) {
                const char* p = (const char*)&ldsA[r * 64] + dc * 64;
                a[dc].q[0] = *(const uint4*)(p + off);
                a[dc].q[1] = *(const uint4*)(p + 32 + off);
            }
        }
        // preload ALL B fragments, then issue the 8 WMMAs (overlap ds/wmma)
        AFrag bf[4][2];
        #pragma unroll
        for (int c = 0; c < 4; ++c) {
            int col = c * 16 + (lane & 15);
            #pragma unroll
            for (int dc = 0; dc < 2; ++dc) {
                const char* p = (const char*)&ldsWt[col * 64] + dc * 64
                                + ((lane < 16) ? 0 : 16);
                bf[c][dc].q[0] = *(const uint4*)p;
                bf[c][dc].q[1] = *(const uint4*)(p + 32);
            }
        }
        #pragma unroll
        for (int c = 0; c < 4; ++c)
            #pragma unroll
            for (int dc = 0; dc < 2; ++dc)
                acc[c] = __builtin_amdgcn_wmma_f32_16x16x32_bf16(
                    false, a[dc].v, false, bf[c][dc].v, (short)0, acc[c], false, false);
    }

    // store: C layout -> bf16 head-split [b,h,l,dep]
    #pragma unroll
    for (int r = 0; r < 8; ++r) {
        int mrow = rowBase + wave * 16 + r + ((lane >> 4) << 3);
        int b = mrow >> 11, l = mrow & (LL - 1);
        #pragma unroll
        for (int c = 0; c < 4; ++c) {
            int n = colBase + c * 16 + (lane & 15);
            int h = n >> 6, dep = n & 63;
            Obf[(((size_t)(b * HH + h)) * LL + l) * DEP + dep] = f2bf(acc[c][r] * scale);
        }
    }
}

// ---------------------------------------------------------------------------
// Flash attention: grid (L/128, B*H), 256 threads (8 waves, 16 q-rows each),
// 64-key blocks, K staged by double-buffered async global->LDS copies.
// Dynamic LDS (bytes): K0 @0 (8K), K1 @8192 (8K), Vt @16384 (8K),
// P @24576 (8 waves x 2K = 16K)  => 40960 total.
// ---------------------------------------------------------------------------
__global__ __launch_bounds__(256) void attn_kernel(const unsigned short* __restrict__ Qbf,
                                                   const unsigned short* __restrict__ Kbf,
                                                   const unsigned short* __restrict__ Vbf,
                                                   const float* __restrict__ bias,
                                                   float* __restrict__ out)
{
    extern __shared__ char smem[];
    unsigned short* ldsK  = (unsigned short*)smem;            // [buf][key][dep]
    unsigned short* ldsVt = (unsigned short*)(smem + 16384);  // [dep][key]
    unsigned short* ldsP  = (unsigned short*)(smem + 24576);  // per-wave 16x64

    const int tid  = threadIdx.x;
    const int lane = tid & 31;
    const int wave = tid >> 5;                   // 0..7
    const int bh = blockIdx.y;                   // b*H + h
    const int b = bh >> 4, h = bh & 15;
    const int qbase = blockIdx.x * 128 + wave * 16;

    const unsigned short* Qrow  = Qbf + ((size_t)bh * LL + qbase) * DEP;
    const unsigned short* Kbase = Kbf + (size_t)bh * LL * DEP;
    const unsigned short* Vbase = Vbf + (size_t)bh * LL * DEP;

    // load Q A-fragments (two depth-chunks of 32)
    AFrag qa[2];
    {
        const char* p = (const char*)(Qrow + (size_t)(lane & 15) * DEP);
        int off = (lane < 16) ? 0 : 16;
        qa[0].q[0] = *(const uint4*)(p + off);
        qa[0].q[1] = *(const uint4*)(p + 32 + off);
        qa[1].q[0] = *(const uint4*)(p + 64 + off);
        qa[1].q[1] = *(const uint4*)(p + 96 + off);
    }

    float m[8], lsum[8];
    v8f acc[4] = {};
    #pragma unroll
    for (int r = 0; r < 8; ++r) { m[r] = -1e30f; lsum[r] = 0.0f; }

    // prologue: async-copy K block 0 into buffer 0 (8KB, 2 x b128 per thread)
    #pragma unroll
    for (int i = 0; i < 2; ++i) {
        int idx = tid + i * 256;
        unsigned off = (unsigned)(idx * 16);
        unsigned long long ga =
            (unsigned long long)(uintptr_t)(Kbase + (size_t)idx * 8);
        asm volatile("global_load_async_to_lds_b128 %0, %1, off"
                     :: "v"(off), "v"(ga) : "memory");
    }

    for (int kb = 0; kb < LL; kb += 64) {
        const int cur = (kb >> 6) & 1;
        __syncthreads();   // all waves done with previous compute / LDS reads

        // issue async copy of NEXT K block into the other buffer
        if (kb + 64 < LL) {
            const unsigned short* g = Kbase + (size_t)(kb + 64) * DEP;
            #pragma unroll
            for (int i = 0; i < 2; ++i) {
                int idx = tid + i * 256;
                unsigned off = (unsigned)((cur ^ 1) * 8192 + idx * 16);
                unsigned long long ga =
                    (unsigned long long)(uintptr_t)(g + (size_t)idx * 8);
                asm volatile("global_load_async_to_lds_b128 %0, %1, off"
                             :: "v"(off), "v"(ga) : "memory");
            }
        }

        // stage V block transposed while the async K copy is in flight:
        // pack key-pairs into u32 -> ds_store_b32
        {
            unsigned* ldsVt32 = (unsigned*)ldsVt;
            int p2 = tid >> 3;                  // key pair: keys 2*p2, 2*p2+1
            int d0 = (tid & 7) * 8;             // 8 deps per thread
            const unsigned short* vp0 = Vbase + (size_t)(kb + p2 * 2) * DEP + d0;
            uint4 a = *(const uint4*)vp0;
            uint4 c = *(const uint4*)(vp0 + DEP);
            const unsigned short* pa8 = (const unsigned short*)&a;
            const unsigned short* pc8 = (const unsigned short*)&c;
            #pragma unroll
            for (int j = 0; j < 8; ++j)
                ldsVt32[(d0 + j) * 32 + p2] =
                    (unsigned)pa8[j] | ((unsigned)pc8[j] << 16);
        }
        if (kb + 64 < LL)
            __builtin_prefetch(Vbase + (size_t)(kb + 64) * DEP + (tid << 4), 0, 0);

        // wait for CURRENT K buffer (async ops complete in order)
        if (kb + 64 < LL) asm volatile("s_wait_asynccnt 0x2" ::: "memory");
        else              asm volatile("s_wait_asynccnt 0x0" ::: "memory");
        __syncthreads();

        const unsigned short* Kblk = ldsK + cur * 4096;   // 4096 ushorts = 8KB

        // S = Q K^T : preload all 8 B-frags, then 8 WMMAs
        AFrag bf[4][2];
        #pragma unroll
        for (int c = 0; c < 4; ++c) {
            int key = c * 16 + (lane & 15);
            #pragma unroll
            for (int dc = 0; dc < 2; ++dc) {
                const char* p = (const char*)&Kblk[key * 64 + dc * 32]
                                + ((lane < 16) ? 0 : 16);
                bf[c][dc].q[0] = *(const uint4*)p;
                bf[c][dc].q[1] = *(const uint4*)(p + 32);
            }
        }
        v8f s[4];
        #pragma unroll
        for (int c = 0; c < 4; ++c) {
            v8f sc = {};
            #pragma unroll
            for (int dc = 0; dc < 2; ++dc)
                sc = __builtin_amdgcn_wmma_f32_16x16x32_bf16(
                    false, qa[dc].v, false, bf[c][dc].v, (short)0, sc, false, false);
            s[c] = sc;
        }

        // + bias[q, k]
        {
            const float* bp = bias + (size_t)qbase * LL + kb;
            #pragma unroll
            for (int r = 0; r < 8; ++r) {
                int row = r + ((lane >> 4) << 3);
                #pragma unroll
                for (int c = 0; c < 4; ++c)
                    s[c][r] += bp[(size_t)row * LL + c * 16 + (lane & 15)];
            }
        }

        // online softmax per row (row spread across a 16-lane group)
        #pragma unroll
        for (int r = 0; r < 8; ++r) {
            float t = fmaxf(fmaxf(s[0][r], s[1][r]), fmaxf(s[2][r], s[3][r]));
            t = fmaxf(t, __shfl_xor(t, 1, 16));
            t = fmaxf(t, __shfl_xor(t, 2, 16));
            t = fmaxf(t, __shfl_xor(t, 4, 16));
            t = fmaxf(t, __shfl_xor(t, 8, 16));
            float mnew = fmaxf(m[r], t);
            float fr = __expf(m[r] - mnew);
            m[r] = mnew;
            int row = r + ((lane >> 4) << 3);
            float ssum = 0.0f;
            #pragma unroll
            for (int c = 0; c < 4; ++c) {
                float pc = __expf(s[c][r] - mnew);
                ldsP[wave * 1024 + row * 64 + c * 16 + (lane & 15)] = f2bf(pc);
                ssum += pc;
            }
            ssum += __shfl_xor(ssum, 1, 16);
            ssum += __shfl_xor(ssum, 2, 16);
            ssum += __shfl_xor(ssum, 4, 16);
            ssum += __shfl_xor(ssum, 8, 16);
            lsum[r] = lsum[r] * fr + ssum;
            #pragma unroll
            for (int c = 0; c < 4; ++c) acc[c][r] *= fr;
        }

        // P A-fragments from this wave's LDS region (16x64 -> two 16x32)
        AFrag pa[2];
        {
            const char* p = (const char*)&ldsP[wave * 1024 + (lane & 15) * 64]
                            + ((lane < 16) ? 0 : 16);
            pa[0].q[0] = *(const uint4*)p;
            pa[0].q[1] = *(const uint4*)(p + 32);
            pa[1].q[0] = *(const uint4*)(p + 64);
            pa[1].q[1] = *(const uint4*)(p + 96);
        }
        // acc += P * V : preload all 8 V B-frags, then 8 WMMAs
        AFrag vb[4][2];
        #pragma unroll
        for (int c = 0; c < 4; ++c) {
            int dep = c * 16 + (lane & 15);
            #pragma unroll
            for (int kc = 0; kc < 2; ++kc) {
                const char* p = (const char*)&ldsVt[dep * 64 + kc * 32]
                                + ((lane < 16) ? 0 : 16);
                vb[c][kc].q[0] = *(const uint4*)p;
                vb[c][kc].q[1] = *(const uint4*)(p + 32);
            }
        }
        #pragma unroll
        for (int c = 0; c < 4; ++c)
            #pragma unroll
            for (int kc = 0; kc < 2; ++kc)
                acc[c] = __builtin_amdgcn_wmma_f32_16x16x32_bf16(
                    false, pa[kc].v, false, vb[c][kc].v, (short)0, acc[c], false, false);
    }

    // epilogue: out[b, l, h*64 + dep] = acc / lsum
    #pragma unroll
    for (int r = 0; r < 8; ++r) {
        float inv = 1.0f / lsum[r];
        int ql = qbase + r + ((lane >> 4) << 3);
        float* op = out + ((size_t)b * LL + ql) * DD + h * DEP;
        #pragma unroll
        for (int c = 0; c < 4; ++c)
            op[c * 16 + (lane & 15)] = acc[c][r] * inv;
    }
}

extern "C" void kernel_launch(void* const* d_in, const int* in_sizes, int n_in,
                              void* d_out, int out_size, void* d_ws, size_t ws_size,
                              hipStream_t stream) {
    (void)in_sizes; (void)n_in; (void)out_size; (void)ws_size;
    const float* x    = (const float*)d_in[0];
    const float* y    = (const float*)d_in[1];
    const float* bias = (const float*)d_in[2];
    const float* Wq   = (const float*)d_in[3];
    const float* Wk   = (const float*)d_in[4];
    const float* Wv   = (const float*)d_in[5];
    float* out = (float*)d_out;

    const size_t perMat = (size_t)BB * HH * LL * DEP;   // 8M bf16 = 16MB
    unsigned short* Qbf = (unsigned short*)d_ws;
    unsigned short* Kbf = Qbf + perMat;
    unsigned short* Vbf = Kbf + perMat;

    dim3 gp((BB * LL) / 128, DD / 64);                  // (64, 16)
    proj_kernel<<<gp, 256, 0, stream>>>(x, Wq, Qbf, 0.125f); // DEPTH^-0.5
    proj_kernel<<<gp, 256, 0, stream>>>(y, Wk, Kbf, 1.0f);
    proj_kernel<<<gp, 256, 0, stream>>>(y, Wv, Vbf, 1.0f);

    dim3 ga(LL / 128, BB * HH);                         // (16, 64)
    attn_kernel<<<ga, 256, 40960, stream>>>(Qbf, Kbf, Vbf, bias, out);
}
